// LSTM_46686294508048
// MI455X (gfx1250) — matrix operand order repaired and checked by reading, not compiled, not run on hardware
//
#include <hip/hip_runtime.h>

typedef __attribute__((ext_vector_type(16))) __bf16 v16bf;
typedef __attribute__((ext_vector_type(8)))  float  v8f;

namespace {
constexpr int kB  = 256;   // batch
constexpr int kT  = 1024;  // timesteps
constexpr int kI  = 64;    // layer-0 input dim
constexpr int kH  = 128;   // hidden dim
constexpr int kG  = 512;   // 4*H gate dim
constexpr int kBT = 16;    // batch tile per workgroup
constexpr int kThreads = 1024; // 32 wave32
constexpr int kW1Elems = 32 * 8 * 32 * 16; // 32 col-tiles x 8 K-frags x 32 lanes x 16
}

__device__ __forceinline__ float sigm(float x) {
    return 1.0f / (1.0f + __expf(-x));
}
__device__ __forceinline__ float tanh_f(float x) {
    float e = __expf(-2.0f * x);
    return (1.0f - e) / (1.0f + e);
}

// Inverse of the A-fragment layout (contiguous K, matching B): given K within a
// 32-wide fragment, return (lane-half, element) position: K = 16*hh + e.
__device__ __forceinline__ void a_pos(int kp, int& hh, int& e) {
    hh = kp >> 4;
    e = kp & 15;
}

__device__ __forceinline__ v8f wmma_bf16(v16bf a, v16bf b, v8f c) {
    // 8 args: (neg_a, A, neg_b, B, c_mod, C, reuse_a, reuse_b)
    return __builtin_amdgcn_wmma_f32_16x16x32_bf16(
        false, a, false, b, (short)0, c, false, false);
}

// Load one K=32 B-fragment slice of a row-major [kG, ldk] weight matrix,
// fp32 -> bf16. B layout: lane n holds column n; lanes 16-31 hold K=16..31.
__device__ __forceinline__ v16bf load_bfrag(const float* __restrict__ W,
                                            int col, int ldk, int kbase, int half) {
    v16bf b;
#pragma unroll
    for (int e = 0; e < 16; ++e)
        b[e] = (__bf16)W[col * ldk + kbase + half * 16 + e];
    return b;
}

__global__ void __launch_bounds__(kThreads, 1)
lstm2_persistent(const float* __restrict__ x,
                 const float* __restrict__ Wih0, const float* __restrict__ Whh0,
                 const float* __restrict__ bih0, const float* __restrict__ bhh0,
                 const float* __restrict__ Wih1, const float* __restrict__ Whh1,
                 const float* __restrict__ bih1, const float* __restrict__ bhh1,
                 const float* __restrict__ Wlin, const float* __restrict__ blin,
                 float* __restrict__ out)
{
    // Layer-1 weights, pre-swizzled into WMMA B-fragment order:
    // frag index ((w*8 + f)*32 + lane)*16 + e ; f=0..3 -> Wih1, f=4..7 -> Whh1.
    __shared__ __align__(32) __bf16 sW1[kW1Elems];      // 256 KB
    // Activations, h pre-swizzled in WMMA A-fragment layout.
    __shared__ __align__(32) __bf16 sXA [2 * 32 * 16];  // x_t  tile, K=64
    __shared__ __align__(32) __bf16 sH0A[4 * 32 * 16];  // h0   tile, K=128
    __shared__ __align__(32) __bf16 sH1A[4 * 32 * 16];  // h1   tile, K=128
    __shared__ float sGates[kBT * kG];                  // 32 KB gate preacts
    __shared__ float sC0[kBT * kH];
    __shared__ float sC1[kBT * kH];

    const int tid  = threadIdx.x;
    const int wave = tid >> 5;
    const int lane = tid & 31;
    const int n    = lane & 15;
    const int half = lane >> 4;
    const int col  = wave * 16 + n;        // gate column owned by this lane
    const int b0   = blockIdx.x * kBT;     // batch-tile origin

    // ---- layer-0 weight slices persistent in VGPRs (6 frags = 48 VGPRs) ----
    v16bf fWih0[2], fWhh0[4];
#pragma unroll
    for (int f = 0; f < 2; ++f) fWih0[f] = load_bfrag(Wih0, col, kI, f * 32, half);
#pragma unroll
    for (int f = 0; f < 4; ++f) fWhh0[f] = load_bfrag(Whh0, col, kH, f * 32, half);

    const float bias0 = bih0[col] + bhh0[col];
    const float bias1 = bih1[col] + bhh1[col];

    // ---- stage layer-1 weights into LDS in B-fragment order (once) ----
    for (int i = tid; i < kW1Elems; i += kThreads) {
        int e  = i & 15;
        int ln = (i >> 4) & 31;
        int f  = (i >> 9) & 7;
        int w  = i >> 12;
        int c_ = w * 16 + (ln & 15);
        int k  = (f & 3) * 32 + (ln >> 4) * 16 + e;
        const float* __restrict__ M = (f < 4) ? Wih1 : Whh1;
        sW1[i] = (__bf16)M[c_ * kH + k];
    }

    // ---- zero initial state ----
    for (int i = tid; i < 4 * 32 * 16; i += kThreads) {
        sH0A[i] = (__bf16)0.0f;
        sH1A[i] = (__bf16)0.0f;
    }
    for (int i = tid; i < kBT * kH; i += kThreads) {
        sC0[i] = 0.0f;
        sC1[i] = 0.0f;
    }
    __syncthreads();

    const __bf16* __restrict__ myW1 = &sW1[(wave * 8) * 32 * 16];

    for (int t = 0; t < kT; ++t) {
        // ---- stage x_t tile (16 x 64 fp32 -> bf16, A-fragment layout) ----
        {
            const int m = tid >> 6;   // batch row in tile
            const int k = tid & 63;   // input feature
            float xv = x[((size_t)(b0 + m) * kT + t) * kI + k];
            int hh, e;
            a_pos(k & 31, hh, e);
            sXA[((k >> 5) * 32 + m + 16 * hh) * 16 + e] = (__bf16)xv;
            if (t + 1 < kT)
                __builtin_prefetch(&x[((size_t)(b0 + m) * kT + t + 1) * kI + k], 0, 1);
        }
        __syncthreads();

        // ---- layer 0 gates: x_t @ Wih0^T + h0 @ Whh0^T + b0 ----
        {
            v8f acc;
#pragma unroll
            for (int r = 0; r < 8; ++r) acc[r] = bias0;
#pragma unroll
            for (int f = 0; f < 2; ++f) {
                v16bf a = *(const v16bf*)&sXA[(f * 32 + lane) * 16];
                acc = wmma_bf16(a, fWih0[f], acc);
            }
#pragma unroll
            for (int f = 0; f < 4; ++f) {
                v16bf a = *(const v16bf*)&sH0A[(f * 32 + lane) * 16];
                acc = wmma_bf16(a, fWhh0[f], acc);
            }
#pragma unroll
            for (int r = 0; r < 8; ++r)
                sGates[(r + 8 * half) * kG + col] = acc[r];
        }
        __syncthreads();

        // ---- layer 0 pointwise: c0,h0 update (i,f,g,o gate order) ----
#pragma unroll
        for (int rep = 0; rep < 2; ++rep) {
            int idx = tid + rep * kThreads;      // 0..2047
            int m   = idx >> 7;
            int hid = idx & 127;
            float gi = sGates[m * kG + hid];
            float gf = sGates[m * kG + 128 + hid];
            float gg = sGates[m * kG + 256 + hid];
            float go = sGates[m * kG + 384 + hid];
            float c  = sigm(gf) * sC0[idx] + sigm(gi) * tanh_f(gg);
            float h  = sigm(go) * tanh_f(c);
            sC0[idx] = c;
            int hh, e;
            a_pos(hid & 31, hh, e);
            sH0A[((hid >> 5) * 32 + m + 16 * hh) * 16 + e] = (__bf16)h;
        }
        __syncthreads();

        // ---- layer 1 gates: h0_t @ Wih1^T + h1 @ Whh1^T + b1 (B from LDS) ----
        {
            v8f acc;
#pragma unroll
            for (int r = 0; r < 8; ++r) acc[r] = bias1;
#pragma unroll
            for (int f = 0; f < 4; ++f) {
                v16bf a = *(const v16bf*)&sH0A[(f * 32 + lane) * 16];
                v16bf b = *(const v16bf*)&myW1[(f * 32 + lane) * 16];
                acc = wmma_bf16(a, b, acc);
            }
#pragma unroll
            for (int f = 0; f < 4; ++f) {
                v16bf a = *(const v16bf*)&sH1A[(f * 32 + lane) * 16];
                v16bf b = *(const v16bf*)&myW1[((4 + f) * 32 + lane) * 16];
                acc = wmma_bf16(a, b, acc);
            }
#pragma unroll
            for (int r = 0; r < 8; ++r)
                sGates[(r + 8 * half) * kG + col] = acc[r];
        }
        __syncthreads();

        // ---- layer 1 pointwise: c1,h1 update ----
#pragma unroll
        for (int rep = 0; rep < 2; ++rep) {
            int idx = tid + rep * kThreads;
            int m   = idx >> 7;
            int hid = idx & 127;
            float gi = sGates[m * kG + hid];
            float gf = sGates[m * kG + 128 + hid];
            float gg = sGates[m * kG + 256 + hid];
            float go = sGates[m * kG + 384 + hid];
            float c  = sigm(gf) * sC1[idx] + sigm(gi) * tanh_f(gg);
            float h  = sigm(go) * tanh_f(c);
            sC1[idx] = c;
            int hh, e;
            a_pos(hid & 31, hh, e);
            sH1A[((hid >> 5) * 32 + m + 16 * hh) * 16 + e] = (__bf16)h;
        }
        __syncthreads();
    }

    // ---- head: out[b] = h1[:, last] . Wlin + blin  (O = 1) ----
    if (tid < kBT) {
        float s = blin[0];
#pragma unroll 4
        for (int hid = 0; hid < kH; ++hid) {
            int hh, e;
            a_pos(hid & 31, hh, e);
            s += (float)sH1A[((hid >> 5) * 32 + tid + 16 * hh) * 16 + e] * Wlin[hid];
        }
        out[b0 + tid] = s;
    }
}

extern "C" void kernel_launch(void* const* d_in, const int* in_sizes, int n_in,
                              void* d_out, int out_size, void* d_ws, size_t ws_size,
                              hipStream_t stream) {
    const float* x    = (const float*)d_in[0];
    const float* Wih0 = (const float*)d_in[1];
    const float* Whh0 = (const float*)d_in[2];
    const float* bih0 = (const float*)d_in[3];
    const float* bhh0 = (const float*)d_in[4];
    const float* Wih1 = (const float*)d_in[5];
    const float* Whh1 = (const float*)d_in[6];
    const float* bih1 = (const float*)d_in[7];
    const float* bhh1 = (const float*)d_in[8];
    const float* Wlin = (const float*)d_in[9];
    const float* blin = (const float*)d_in[10];
    float* out = (float*)d_out;

    lstm2_persistent<<<dim3(kB / kBT), dim3(kThreads), 0, stream>>>(
        x, Wih0, Whh0, bih0, bhh0, Wih1, Whh1, bih1, bhh1, Wlin, blin, out);
}
